// DiTBlock_34668976013505
// MI455X (gfx1250) — matrix operand order repaired and checked by reading, not compiled, not run on hardware
//
#include <hip/hip_runtime.h>
#include <math.h>

// ---------- CDNA5 WMMA types ----------
typedef __attribute__((ext_vector_type(16))) __bf16 v16bf;
typedef __attribute__((ext_vector_type(8)))  __bf16 v8bf;
typedef __attribute__((ext_vector_type(8)))  float  v8f;

// ---------- CDNA5 async copy: global -> LDS (ASYNCcnt pipe) ----------
// Per-lane: 16 bytes from gaddr to the lane's LDS offset. LDS byte offset is
// the low 32 bits of the flat shared-aperture address (ISA 10.2).
__device__ __forceinline__ void async_copy_b128(const void* gaddr, void* lds) {
  unsigned lds_off = (unsigned)(size_t)lds;
  asm volatile("global_load_async_to_lds_b128 %0, %1, off"
               :: "v"(lds_off), "v"((unsigned long long)(size_t)gaddr)
               : "memory");
}
__device__ __forceinline__ void wait_async0() {
  asm volatile("s_wait_asynccnt 0" ::: "memory");
}

// =====================================================================
// Elementwise helpers
// =====================================================================
__global__ void f32_to_bf16_v4(const float* __restrict__ x, __bf16* __restrict__ y, long n4) {
  long i = (long)blockIdx.x * blockDim.x + threadIdx.x;
  if (i >= n4) return;
  float4 v = ((const float4*)x)[i];
  __bf16 o[4] = { (__bf16)v.x, (__bf16)v.y, (__bf16)v.z, (__bf16)v.w };
  ((uint2*)y)[i] = *(uint2*)o;
}

__global__ void copy_f32_v4(const float* __restrict__ x, float* __restrict__ y, long n4) {
  long i = (long)blockIdx.x * blockDim.x + threadIdx.x;
  if (i >= n4) return;
  ((float4*)y)[i] = ((const float4*)x)[i];
}

// =====================================================================
// LayerNorm (eps 1e-6, no affine): fp32 row -> bf16 row.  C == 1024,
// block = 256 threads (8 wave32), one row per block.
// =====================================================================
__device__ __forceinline__ float block_sum_256(float v, float* red) {
#pragma unroll
  for (int m = 16; m; m >>= 1) v += __shfl_xor(v, m, 32);
  const int lane = threadIdx.x & 31, w = threadIdx.x >> 5;
  if (lane == 0) red[w] = v;
  __syncthreads();
  float t = (lane < 8) ? red[lane] : 0.0f;
#pragma unroll
  for (int m = 4; m; m >>= 1) t += __shfl_xor(t, m, 32);
  t = __shfl(t, 0, 32);
  __syncthreads();
  return t;
}

__global__ __launch_bounds__(256)
void layernorm_to_bf16(const float* __restrict__ x, __bf16* __restrict__ h, int C) {
  __shared__ float red[8];
  const int row = blockIdx.x;
  const float* xr = x + (size_t)row * C;
  float4 v = ((const float4*)xr)[threadIdx.x];
  const float invC = 1.0f / (float)C;
  float mu = block_sum_256(v.x + v.y + v.z + v.w, red) * invC;
  float dx = v.x - mu, dy = v.y - mu, dz = v.z - mu, dw = v.w - mu;
  float var = block_sum_256(dx*dx + dy*dy + dz*dz + dw*dw, red) * invC;
  float rs = rsqrtf(var + 1e-6f);
  __bf16 o[4] = { (__bf16)(dx*rs), (__bf16)(dy*rs), (__bf16)(dz*rs), (__bf16)(dw*rs) };
  ((uint2*)(h + (size_t)row * C))[threadIdx.x] = *(uint2*)o;
}

// =====================================================================
// bf16 WMMA GEMM:  out[M,N] = A[M,K] @ B[K,N] + bias (+res) (gelu?)
// Tile: BM=128, BN=128, BK=32. 256 threads = 8 waves in a 4(M) x 2(N)
// grid; each wave owns a 32x64 patch = 2x4 v_wmma_f32_16x16x32_bf16.
// A tile staged with global_load_async_to_lds_b128 (ASYNCcnt); B tile
// transposed through VGPRs (async copy cannot transpose).
// =====================================================================
template<bool OUT_BF16, bool GELU_EPI, bool RES_EPI>
__global__ __launch_bounds__(256)
void gemm_bf16_wmma(const __bf16* __restrict__ A, int lda,
                    const __bf16* __restrict__ Bm, int ldb,
                    const float* __restrict__ bias,
                    const float* __restrict__ res, int ldr,
                    void* __restrict__ out, int ldo,
                    int M, int N, int K) {
  constexpr int BM = 128, BN = 128, BK = 32, PAD = 8;
  __shared__ __bf16 sA[BM][BK + PAD];   // row-major [row][k]
  __shared__ __bf16 sBt[BN][BK + PAD];  // transposed [n][k]

  const int tid  = threadIdx.x;
  const int lane = tid & 31, wave = tid >> 5;
  const int wm = wave & 3, wn = wave >> 2;
  const int half = lane >> 4, l16 = lane & 15;
  const int m0 = blockIdx.y * BM;
  const int n0 = blockIdx.x * BN;

  v8f acc[2][4] = {};

  for (int k0 = 0; k0 < K; k0 += BK) {
    // ---- A tile 128x32: async copy, 16B per lane, 2 rounds
#pragma unroll
    for (int r = 0; r < 2; ++r) {
      int idx = (r * 256 + tid) * 8;
      int row = idx >> 5;
      int col = idx & 31;
      async_copy_b128(A + (size_t)(m0 + row) * lda + k0 + col, &sA[row][col]);
    }
    // ---- B tile 32x128 -> transposed store, 2 rounds of 8 bf16
#pragma unroll
    for (int r = 0; r < 2; ++r) {
      int idx  = (r * 256 + tid) * 8;
      int krow = idx >> 7;
      int ncol = idx & 127;
      uint4 d = *(const uint4*)(Bm + (size_t)(k0 + krow) * ldb + n0 + ncol);
      __bf16 e[8]; *(uint4*)e = d;
#pragma unroll
      for (int j = 0; j < 8; ++j) sBt[ncol + j][krow] = e[j];
    }
    wait_async0();
    __syncthreads();

    if (k0 + BK < K) {  // gfx1250 global_prefetch_b8 of next tiles
      __builtin_prefetch(A + (size_t)(m0 + (tid >> 1)) * lda + k0 + BK, 0, 1);
      __builtin_prefetch(Bm + (size_t)(k0 + BK + (tid >> 3)) * ldb + n0, 0, 1);
    }

    // ---- fragments (A: lane=row, K halves 0..7/16..23 vs 8..15/24..31;
    //                 B: lane=col, same K halves as A)
    v16bf afrag[2], bfrag[4];
#pragma unroll
    for (int mt = 0; mt < 2; ++mt) {
      int row = wm * 32 + mt * 16 + l16;
      *((v8bf*)&afrag[mt])     = *(const v8bf*)&sA[row][half * 8];
      *((v8bf*)&afrag[mt] + 1) = *(const v8bf*)&sA[row][16 + half * 8];
    }
#pragma unroll
    for (int nt = 0; nt < 4; ++nt) {
      int coln = wn * 64 + nt * 16 + l16;
      *((v8bf*)&bfrag[nt])     = *(const v8bf*)&sBt[coln][half * 8];
      *((v8bf*)&bfrag[nt] + 1) = *(const v8bf*)&sBt[coln][16 + half * 8];
    }
#pragma unroll
    for (int mt = 0; mt < 2; ++mt)
#pragma unroll
      for (int nt = 0; nt < 4; ++nt)
        acc[mt][nt] = __builtin_amdgcn_wmma_f32_16x16x32_bf16(
            false, afrag[mt], false, bfrag[nt], (short)0, acc[mt][nt], false, false);
    __syncthreads();
  }

  // ---- epilogue
#pragma unroll
  for (int mt = 0; mt < 2; ++mt) {
#pragma unroll
    for (int nt = 0; nt < 4; ++nt) {
      int col = n0 + wn * 64 + nt * 16 + l16;
      float bv = bias[col];
#pragma unroll
      for (int r = 0; r < 8; ++r) {
        int row = m0 + wm * 32 + mt * 16 + r + half * 8;
        float v = acc[mt][nt][r] + bv;
        if (GELU_EPI) v = 0.5f * v * (1.0f + erff(v * 0.70710678118654752f));
        if (RES_EPI)  v += res[(size_t)row * ldr + col];
        if (OUT_BF16) ((__bf16*)out)[(size_t)row * ldo + col] = (__bf16)v;
        else          ((float*)out)[(size_t)row * ldo + col] = v;
      }
    }
  }
}

// =====================================================================
// Flash attention, bf16 WMMA. qkv: [B*N, 3C] bf16 (q|k|v slabs).
// Block = 128 threads (4 waves); wave handles 16 query rows; block
// handles 64 rows of one (b,h). Streams keys/values in chunks of 32
// with online softmax (mask-additive -1e4, scale D^-0.5). K tile is
// staged with async copies; V^T through VGPRs.
// =====================================================================
__global__ __launch_bounds__(128)
void attn_flash_wmma(const __bf16* __restrict__ qkv,
                     const int* __restrict__ mask,
                     __bf16* __restrict__ outa,
                     int N, int C, int H) {
  const int D = C / H;                 // 64
  const int b = blockIdx.y / H, h = blockIdx.y % H;
  const int tid = threadIdx.x, lane = tid & 31, wave = tid >> 5;
  const int half = lane >> 4, l16 = lane & 15;
  const float scale = rsqrtf((float)D);

  const size_t rs3 = (size_t)3 * C;
  const __bf16* qbase = qkv + (size_t)b * N * rs3 + (size_t)h * D;
  const __bf16* kbase = qbase + C;
  const __bf16* vbase = qbase + 2 * C;

  __shared__ __bf16 sK[32][64 + 8];        // [kcol][d]
  __shared__ __bf16 sVt[64][32 + 8];       // [d][kcol]
  __shared__ __bf16 sP[4][16][32 + 8];     // per-wave P staging

  // Q fragments for this wave's 16 rows, full D=64 (two K=32 chunks)
  const int qr = blockIdx.x * 64 + wave * 16;
  v16bf qfrag[2];
  {
    const __bf16* qrow = qbase + (size_t)(qr + l16) * rs3;
#pragma unroll
    for (int kk = 0; kk < 2; ++kk) {
      *((v8bf*)&qfrag[kk])     = *(const v8bf*)(qrow + kk * 32 + half * 8);
      *((v8bf*)&qfrag[kk] + 1) = *(const v8bf*)(qrow + kk * 32 + 16 + half * 8);
    }
  }

  v8f o[4] = {};
  float mrow[8], lrow[8];
#pragma unroll
  for (int r = 0; r < 8; ++r) { mrow[r] = -3.0e38f; lrow[r] = 0.0f; }

  for (int j0 = 0; j0 < N; j0 += 32) {
    // ---- stage K chunk [32][64] via async copy; V^T [64][32] via VGPRs
    {
      int kr = tid >> 2;               // 0..31
      int dc = (tid & 3) * 16;         // 0,16,32,48
      const __bf16* ks = kbase + (size_t)(j0 + kr) * rs3 + dc;
      async_copy_b128(ks,     &sK[kr][dc]);
      async_copy_b128(ks + 8, &sK[kr][dc + 8]);
      const __bf16* vs = vbase + (size_t)(j0 + kr) * rs3 + dc;
      uint4 b0 = *(const uint4*)vs;
      uint4 b1 = *(const uint4*)(vs + 8);
      __bf16 tv[16]; *(uint4*)&tv[0] = b0; *(uint4*)&tv[8] = b1;
#pragma unroll
      for (int j = 0; j < 16; ++j) sVt[dc + j][kr] = tv[j];
    }
    wait_async0();
    __syncthreads();

    // ---- S = Q K^T (two 16x16 tiles), scale + mask
    v8f s[2];
#pragma unroll
    for (int jt = 0; jt < 2; ++jt) {
      int kc = jt * 16 + l16;
      v16bf bk[2];
#pragma unroll
      for (int kk = 0; kk < 2; ++kk) {
        *((v8bf*)&bk[kk])     = *(const v8bf*)&sK[kc][kk * 32 + half * 8];
        *((v8bf*)&bk[kk] + 1) = *(const v8bf*)&sK[kc][kk * 32 + 16 + half * 8];
      }
      v8f t = {};
      t = __builtin_amdgcn_wmma_f32_16x16x32_bf16(false, qfrag[0], false, bk[0], (short)0, t, false, false);
      t = __builtin_amdgcn_wmma_f32_16x16x32_bf16(false, qfrag[1], false, bk[1], (short)0, t, false, false);
      int kcol = j0 + jt * 16 + l16;
      float madd = (mask[(size_t)b * N + kcol] != 1) ? -10000.0f : 0.0f;
#pragma unroll
      for (int r = 0; r < 8; ++r) s[jt][r] = t[r] * scale + madd;
    }

    // ---- online softmax (row stats live in each 16-lane half)
    float mnew[8], alpha[8];
#pragma unroll
    for (int r = 0; r < 8; ++r) {
      float mx = fmaxf(s[0][r], s[1][r]);
#pragma unroll
      for (int sh = 8; sh; sh >>= 1) mx = fmaxf(mx, __shfl_xor(mx, sh, 32));
      mnew[r]  = fmaxf(mrow[r], mx);
      alpha[r] = __expf(mrow[r] - mnew[r]);
      mrow[r]  = mnew[r];
    }
#pragma unroll
    for (int jt = 0; jt < 2; ++jt)
#pragma unroll
      for (int r = 0; r < 8; ++r) s[jt][r] = __expf(s[jt][r] - mnew[r]);
#pragma unroll
    for (int r = 0; r < 8; ++r) {
      float ps = s[0][r] + s[1][r];
#pragma unroll
      for (int sh = 8; sh; sh >>= 1) ps += __shfl_xor(ps, sh, 32);
      lrow[r] = lrow[r] * alpha[r] + ps;
    }
#pragma unroll
    for (int n = 0; n < 4; ++n)
#pragma unroll
      for (int r = 0; r < 8; ++r) o[n][r] *= alpha[r];

    // ---- restage P (C-layout -> A-layout) through per-wave LDS
#pragma unroll
    for (int jt = 0; jt < 2; ++jt)
#pragma unroll
      for (int r = 0; r < 8; ++r)
        sP[wave][r + half * 8][jt * 16 + l16] = (__bf16)s[jt][r];

    v16bf pfrag;
    *((v8bf*)&pfrag)     = *(const v8bf*)&sP[wave][l16][half * 8];
    *((v8bf*)&pfrag + 1) = *(const v8bf*)&sP[wave][l16][16 + half * 8];

    // ---- O += P V
#pragma unroll
    for (int n = 0; n < 4; ++n) {
      int dcol = n * 16 + l16;
      v16bf bv;
      *((v8bf*)&bv)     = *(const v8bf*)&sVt[dcol][half * 8];
      *((v8bf*)&bv + 1) = *(const v8bf*)&sVt[dcol][16 + half * 8];
      o[n] = __builtin_amdgcn_wmma_f32_16x16x32_bf16(false, pfrag, false, bv, (short)0, o[n], false, false);
    }
    __syncthreads();
  }

  // ---- finalize and store bf16 [B*N, C]
#pragma unroll
  for (int n = 0; n < 4; ++n) {
    int dcol = h * D + n * 16 + l16;
#pragma unroll
    for (int r = 0; r < 8; ++r) {
      int row = qr + r + half * 8;
      float val = o[n][r] / lrow[r];
      outa[(size_t)(b * N + row) * C + dcol] = (__bf16)val;
    }
  }
}

// =====================================================================
// Launch
// =====================================================================
extern "C" void kernel_launch(void* const* d_in, const int* in_sizes, int n_in,
                              void* d_out, int out_size, void* d_ws, size_t ws_size,
                              hipStream_t stream) {
  (void)in_sizes; (void)n_in; (void)out_size; (void)ws_size;
  const int B = 4, N = 1024, C = 1024, H = 16;
  const int M = B * N, C3 = 3 * C, MH = 4 * C;

  const float* x        = (const float*)d_in[0];
  const float* c        = (const float*)d_in[1];
  const int*   mask     = (const int*)d_in[2];
  const float* sa_qkv_w = (const float*)d_in[3];
  const float* sa_qkv_b = (const float*)d_in[4];
  const float* sa_pj_w  = (const float*)d_in[5];
  const float* sa_pj_b  = (const float*)d_in[6];
  const float* ca_q_w   = (const float*)d_in[7];
  const float* ca_q_b   = (const float*)d_in[8];
  const float* ca_k_w   = (const float*)d_in[9];
  const float* ca_k_b   = (const float*)d_in[10];
  const float* ca_v_w   = (const float*)d_in[11];
  const float* ca_v_b   = (const float*)d_in[12];
  const float* ca_pj_w  = (const float*)d_in[13];
  const float* ca_pj_b  = (const float*)d_in[14];
  const float* fc1_w    = (const float*)d_in[15];
  const float* fc1_b    = (const float*)d_in[16];
  const float* fc2_w    = (const float*)d_in[17];
  const float* fc2_b    = (const float*)d_in[18];

  char* ws = (char*)d_ws;
  size_t off = 0;
  auto alloc = [&](size_t bytes) -> void* {
    void* p = ws + off;
    off = (off + bytes + 255) & ~(size_t)255;
    return p;
  };
  __bf16* wqkv = (__bf16*)alloc((size_t)C * C3 * 2);
  __bf16* wsp  = (__bf16*)alloc((size_t)C * C * 2);
  __bf16* wcq  = (__bf16*)alloc((size_t)C * C * 2);
  __bf16* wck  = (__bf16*)alloc((size_t)C * C * 2);
  __bf16* wcv  = (__bf16*)alloc((size_t)C * C * 2);
  __bf16* wcp  = (__bf16*)alloc((size_t)C * C * 2);
  __bf16* wf1  = (__bf16*)alloc((size_t)C * MH * 2);
  __bf16* wf2  = (__bf16*)alloc((size_t)MH * C * 2);
  __bf16* cbf  = (__bf16*)alloc((size_t)M * C * 2);
  __bf16* hbf  = (__bf16*)alloc((size_t)M * C * 2);
  __bf16* qkvb = (__bf16*)alloc((size_t)M * C3 * 2);
  __bf16* abf  = (__bf16*)alloc((size_t)M * C * 2);
  __bf16* mlpb = (__bf16*)alloc((size_t)M * MH * 2);
  float*  x0   = (float*)alloc((size_t)M * C * 4);
  float*  x1   = (float*)alloc((size_t)M * C * 4);

  auto conv = [&](const float* s, __bf16* d, long n) {
    long n4 = n / 4;
    f32_to_bf16_v4<<<(unsigned)((n4 + 255) / 256), 256, 0, stream>>>(s, d, n4);
  };
  conv(sa_qkv_w, wqkv, (long)C * C3);
  conv(sa_pj_w,  wsp,  (long)C * C);
  conv(ca_q_w,   wcq,  (long)C * C);
  conv(ca_k_w,   wck,  (long)C * C);
  conv(ca_v_w,   wcv,  (long)C * C);
  conv(ca_pj_w,  wcp,  (long)C * C);
  conv(fc1_w,    wf1,  (long)C * MH);
  conv(fc2_w,    wf2,  (long)MH * C);
  conv(c,        cbf,  (long)M * C);
  {
    long n4 = (long)M * C / 4;
    copy_f32_v4<<<(unsigned)((n4 + 255) / 256), 256, 0, stream>>>(x, x0, n4);
  }

  const dim3 blk(256);
  const dim3 gAttn(N / 64, B * H);

  // ---------------- self-attention ----------------
  layernorm_to_bf16<<<M, 256, 0, stream>>>(x0, hbf, C);
  gemm_bf16_wmma<true, false, false><<<dim3(C3 / 128, M / 128), blk, 0, stream>>>(
      hbf, C, wqkv, C3, sa_qkv_b, nullptr, 0, (void*)qkvb, C3, M, C3, C);
  attn_flash_wmma<<<gAttn, 128, 0, stream>>>(qkvb, mask, abf, N, C, H);
  gemm_bf16_wmma<false, false, true><<<dim3(C / 128, M / 128), blk, 0, stream>>>(
      abf, C, wsp, C, sa_pj_b, x0, C, (void*)x1, C, M, C, C);

  // ---------------- cross-attention ----------------
  layernorm_to_bf16<<<M, 256, 0, stream>>>(x1, hbf, C);
  gemm_bf16_wmma<true, false, false><<<dim3(C / 128, M / 128), blk, 0, stream>>>(
      hbf, C, wcq, C, ca_q_b, nullptr, 0, (void*)qkvb, C3, M, C, C);
  gemm_bf16_wmma<true, false, false><<<dim3(C / 128, M / 128), blk, 0, stream>>>(
      cbf, C, wck, C, ca_k_b, nullptr, 0, (void*)(qkvb + C), C3, M, C, C);
  gemm_bf16_wmma<true, false, false><<<dim3(C / 128, M / 128), blk, 0, stream>>>(
      cbf, C, wcv, C, ca_v_b, nullptr, 0, (void*)(qkvb + 2 * C), C3, M, C, C);
  attn_flash_wmma<<<gAttn, 128, 0, stream>>>(qkvb, mask, abf, N, C, H);
  gemm_bf16_wmma<false, false, true><<<dim3(C / 128, M / 128), blk, 0, stream>>>(
      abf, C, wcp, C, ca_pj_b, x1, C, (void*)x0, C, M, C, C);

  // ---------------- MLP ----------------
  layernorm_to_bf16<<<M, 256, 0, stream>>>(x0, hbf, C);
  gemm_bf16_wmma<true, true, false><<<dim3(MH / 128, M / 128), blk, 0, stream>>>(
      hbf, C, wf1, MH, fc1_b, nullptr, 0, (void*)mlpb, MH, M, MH, C);
  gemm_bf16_wmma<false, false, true><<<dim3(C / 128, M / 128), blk, 0, stream>>>(
      mlpb, MH, wf2, C, fc2_b, x0, C, d_out, C, M, C, MH);
}
